// SHMA_19593640804440
// MI455X (gfx1250) — compile-verified, matrix operand order checked by reading, not run on hardware
//
#include <hip/hip_runtime.h>
#include <hip/hip_bf16.h>

// ---------------------------------------------------------------------------
// Types
// ---------------------------------------------------------------------------
typedef __bf16        v16bf __attribute__((ext_vector_type(16)));
typedef float         v8f   __attribute__((ext_vector_type(8)));
typedef unsigned int  u32x4 __attribute__((ext_vector_type(4)));
typedef float         f32x4 __attribute__((ext_vector_type(4)));

union AFrag { v16bf v; u32x4 q[2]; };

// Problem constants
static constexpr int BATCH = 8;
static constexpr int DIM   = 512;   // channels / d
static constexpr int NPix  = 4096;  // H*W
static constexpr int M4    = 2048;  // 4*d

// ---------------------------------------------------------------------------
// CDNA5 async global->LDS copy (GLOBAL_LOAD_ASYNC_TO_LDS_B128, ASYNCcnt).
// A generic pointer to __shared__ carries the LDS byte offset in its low 32
// bits (AS3->flat addrspacecast = {aperture_hi32, lds_offset32}).
// ---------------------------------------------------------------------------
__device__ __forceinline__ void async_cp16(const __bf16* g, __bf16* l)
{
    unsigned lds = (unsigned)(unsigned long long)l;
    asm volatile("global_load_async_to_lds_b128 %0, %1, off"
                 :: "v"(lds), "v"(g) : "memory");
}

template <int N>
__device__ __forceinline__ void wait_async()
{
    asm volatile("s_wait_asynccnt %0" :: "n"(N) : "memory");
}

// Issue one 128x32 bf16 tile of A and one of B (4 async instructions / wave).
__device__ __forceinline__ void issue_tile(
    const __bf16* __restrict__ gA, const __bf16* __restrict__ gB,
    __bf16* lA, __bf16* lB, int tid, int K)
{
#pragma unroll
    for (int it = 0; it < 2; ++it) {
        int p   = tid + it * 256;
        int row = p >> 2;
        int ch  = p & 3;
        async_cp16(gA + (size_t)row * K + ch * 8, lA + row * 32 + ch * 8);
        async_cp16(gB + (size_t)row * K + ch * 8, lB + row * 32 + ch * 8);
    }
}

// ---------------------------------------------------------------------------
// Tiled bf16 WMMA GEMM:  Out(M x N) = A(M x K, row major) @ Bt(N x K)^T
// Block tile 128x128, K-step 32, 256 threads = 8 waves (2 in M x 4 in N),
// each wave computes 64x32 via 4x2 fragments of v_wmma_f32_16x16x32_bf16.
// Double-buffered LDS, software-pipelined with async copies:
//   iter t: barrier; issue tile t+1 -> buf[(t+1)&1]; s_wait_asynccnt<=4
//           (tile t resident); barrier; 8x WMMA from buf[t&1].
// MODE 0: write bf16, row major (ldOut = N).  MODE 1: write fp32 transposed
// (Out[n][m], ldOut = M).
// ---------------------------------------------------------------------------
template <int MODE>
__global__ __launch_bounds__(256) void gemm_bf16_wmma(
    const __bf16* __restrict__ A,  long long strideA,
    const __bf16* __restrict__ Bt, long long strideB,
    void* __restrict__ Out,        long long strideOut,
    int K, int ldOut, float scale)
{
    __shared__ __bf16 sA[2][128 * 32];
    __shared__ __bf16 sB[2][128 * 32];

    const int tid  = threadIdx.x;
    const int b    = blockIdx.z;
    const int m0   = blockIdx.y * 128;
    const int n0   = blockIdx.x * 128;
    const int w    = tid >> 5;
    const int lane = tid & 31;
    const int m16  = lane & 15;
    const int hi   = lane >> 4;
    const int wm   = w & 1;   // 2 waves along M
    const int wn   = w >> 1;  // 4 waves along N

    const __bf16* Ab = A  + (size_t)b * strideA + (size_t)m0 * K;
    const __bf16* Bb = Bt + (size_t)b * strideB + (size_t)n0 * K;

    v8f acc[4][2];
    const v8f vzero = {0.f, 0.f, 0.f, 0.f, 0.f, 0.f, 0.f, 0.f};
#pragma unroll
    for (int mt = 0; mt < 4; ++mt)
#pragma unroll
        for (int nt = 0; nt < 2; ++nt) acc[mt][nt] = vzero;

    const int nT = K >> 5;
    // Prologue: tile 0 -> buffer 0.
    issue_tile(Ab, Bb, sA[0], sB[0], tid, K);

    for (int t = 0; t < nT; ++t) {
        __syncthreads();  // prior reads of buf[(t+1)&1] are done in all waves
        if (t + 1 < nT) {
            issue_tile(Ab + (t + 1) * 32, Bb + (t + 1) * 32,
                       sA[(t + 1) & 1], sB[(t + 1) & 1], tid, K);
            wait_async<4>();   // 4 just issued remain; tile t is resident
        } else {
            wait_async<0>();
        }
        __syncthreads();  // tile t visible to all waves

        const __bf16* cA = sA[t & 1];
        const __bf16* cB = sB[t & 1];

        // Fragment loads per documented 16-bit layout: lane(m16,hi), elem j ->
        // K = (j<8 ? j : j+8) + 8*hi  => two contiguous 16B LDS reads.
        AFrag af[4];
#pragma unroll
        for (int mt = 0; mt < 4; ++mt) {
            int row = wm * 64 + mt * 16 + m16;
            af[mt].q[0] = *(const u32x4*)(cA + row * 32 + hi * 8);
            af[mt].q[1] = *(const u32x4*)(cA + row * 32 + 16 + hi * 8);
        }
        AFrag bfr[2];
#pragma unroll
        for (int nt = 0; nt < 2; ++nt) {
            int row = wn * 32 + nt * 16 + m16;
            bfr[nt].q[0] = *(const u32x4*)(cB + row * 32 + hi * 8);
            bfr[nt].q[1] = *(const u32x4*)(cB + row * 32 + 16 + hi * 8);
        }
#pragma unroll
        for (int mt = 0; mt < 4; ++mt)
#pragma unroll
            for (int nt = 0; nt < 2; ++nt)
                acc[mt][nt] = __builtin_amdgcn_wmma_f32_16x16x32_bf16(
                    false, af[mt].v, false, bfr[nt].v,
                    (short)0, acc[mt][nt], false, false);
    }

    if (MODE == 0) {
        __bf16* O = (__bf16*)Out + (size_t)b * strideOut;
#pragma unroll
        for (int mt = 0; mt < 4; ++mt)
#pragma unroll
            for (int nt = 0; nt < 2; ++nt) {
                int gn = n0 + wn * 32 + nt * 16 + m16;
#pragma unroll
                for (int r = 0; r < 8; ++r) {
                    int gm = m0 + wm * 64 + mt * 16 + r + 8 * hi;
                    O[(size_t)gm * ldOut + gn] = (__bf16)(acc[mt][nt][r] * scale);
                }
            }
    } else {
        // Transposed fp32 store: lane's 8 accum regs are 8 consecutive M.
        float* O = (float*)Out + (size_t)b * strideOut;
#pragma unroll
        for (int mt = 0; mt < 4; ++mt)
#pragma unroll
            for (int nt = 0; nt < 2; ++nt) {
                int gn  = n0 + wn * 32 + nt * 16 + m16;
                int gmb = m0 + wm * 64 + mt * 16 + 8 * hi;
                f32x4 lo = {acc[mt][nt][0] * scale, acc[mt][nt][1] * scale,
                            acc[mt][nt][2] * scale, acc[mt][nt][3] * scale};
                f32x4 hi4 = {acc[mt][nt][4] * scale, acc[mt][nt][5] * scale,
                             acc[mt][nt][6] * scale, acc[mt][nt][7] * scale};
                *(f32x4*)(O + (size_t)gn * ldOut + gmb)     = lo;
                *(f32x4*)(O + (size_t)gn * ldOut + gmb + 4) = hi4;
            }
    }
}

// ---------------------------------------------------------------------------
// fp32 -> bf16 copy
// ---------------------------------------------------------------------------
__global__ void f2bf_kernel(const float* __restrict__ src, __bf16* __restrict__ dst, int n)
{
    int i = blockIdx.x * 256 + threadIdx.x;
    if (i < n) dst[i] = (__bf16)src[i];
}

// ---------------------------------------------------------------------------
// x (B,512,4096) fp32 -> xbT (B,4096,512) bf16  (tiled LDS transpose)
// ---------------------------------------------------------------------------
__global__ __launch_bounds__(256) void transpose_x_kernel(
    const float* __restrict__ x, __bf16* __restrict__ xbT)
{
    __shared__ float tile[32][33];
    const int b  = blockIdx.z;
    const int c0 = blockIdx.y * 32;
    const int n0 = blockIdx.x * 32;
    const int col = threadIdx.x & 31;
    const int row = threadIdx.x >> 5; // 0..7
#pragma unroll
    for (int rr = 0; rr < 4; ++rr) {
        int r = row + rr * 8;
        tile[r][col] = x[((size_t)b * DIM + c0 + r) * NPix + n0 + col];
    }
    __syncthreads();
#pragma unroll
    for (int rr = 0; rr < 4; ++rr) {
        int r = row + rr * 8;
        xbT[((size_t)b * NPix + n0 + r) * DIM + c0 + col] = (__bf16)tile[col][r];
    }
}

// ---------------------------------------------------------------------------
// Per-channel BN stats (sum, sumsq) over B x N, deterministic reduction.
// ---------------------------------------------------------------------------
__global__ __launch_bounds__(256) void bn_stats_kernel(
    const __bf16* __restrict__ Y, float* __restrict__ stats, long long batchStride)
{
    __shared__ float rs[256];
    __shared__ float rq[256];
    const int ch = blockIdx.x;
    float s = 0.f, q = 0.f;
    for (int b = 0; b < BATCH; ++b) {
        const __bf16* p = Y + (size_t)b * batchStride + (size_t)ch * NPix;
        for (int n = threadIdx.x; n < NPix; n += 256) {
            float v = (float)p[n];
            s += v; q += v * v;
        }
    }
    int t = threadIdx.x;
    rs[t] = s; rq[t] = q;
    __syncthreads();
    for (int st = 128; st > 0; st >>= 1) {
        if (t < st) { rs[t] += rs[t + st]; rq[t] += rq[t + st]; }
        __syncthreads();
    }
    if (t == 0) { stats[2 * ch] = rs[0]; stats[2 * ch + 1] = rq[0]; }
}

__global__ void bn_finalize_kernel(
    const float* __restrict__ stats, const float* __restrict__ gamma,
    const float* __restrict__ beta, float* __restrict__ coef, int Mch, float invCnt)
{
    int t = blockIdx.x * 256 + threadIdx.x;
    if (t < Mch) {
        float mean = stats[2 * t] * invCnt;
        float var  = stats[2 * t + 1] * invCnt - mean * mean;
        float inv  = rsqrtf(var + 1e-5f);
        float a    = gamma[t] * inv;
        coef[2 * t]     = a;
        coef[2 * t + 1] = beta[t] - mean * a;
    }
}

// ---------------------------------------------------------------------------
// Apply BN + activations; scatter into layouts needed by the next GEMMs.
// quad 0 (q): sigmoid -> qbT (B,N,512);  quad 1 (k): sigmoid -> kb (B,512,N)
// + k-mean; quad 2 (v): identity -> vb (B,512,N); quad 3 (g): silu -> gT (B,N,512)
// ---------------------------------------------------------------------------
__global__ __launch_bounds__(256) void act_kernel(
    const __bf16* __restrict__ Y, const float* __restrict__ coef,
    __bf16* __restrict__ qbT, __bf16* __restrict__ kb,
    __bf16* __restrict__ vb,  __bf16* __restrict__ gT,
    float* __restrict__ kmean)
{
    __shared__ float red[256];
    const int m4   = blockIdx.x;
    const int b    = blockIdx.y;
    const int quad = m4 >> 9;
    const int ch   = m4 & 511;
    const float a  = coef[2 * m4];
    const float bb = coef[2 * m4 + 1];
    const __bf16* yrow = Y + ((size_t)b * M4 + m4) * NPix;

    float ksum = 0.f;
    for (int i = 0; i < 16; ++i) {
        int n = threadIdx.x + i * 256;
        float v = a * (float)yrow[n] + bb;
        if (quad == 0) {
            float s = 1.f / (1.f + __expf(-v));
            qbT[((size_t)b * NPix + n) * DIM + ch] = (__bf16)s;
        } else if (quad == 1) {
            float s = 1.f / (1.f + __expf(-v));
            kb[((size_t)b * DIM + ch) * NPix + n] = (__bf16)s;
            ksum += s;
        } else if (quad == 2) {
            vb[((size_t)b * DIM + ch) * NPix + n] = (__bf16)v;
        } else {
            float s = v / (1.f + __expf(-v));
            gT[((size_t)b * NPix + n) * DIM + ch] = (__bf16)s;
        }
    }
    if (quad == 1) {
        int t = threadIdx.x;
        red[t] = ksum;
        __syncthreads();
        for (int st = 128; st > 0; st >>= 1) {
            if (t < st) red[t] += red[t + st];
            __syncthreads();
        }
        if (t == 0) kmean[(size_t)b * DIM + ch] = red[0] * (1.f / (float)NPix);
    }
}

// ---------------------------------------------------------------------------
// z[b,n] = dot(kmean[b,:], qbT[b,n,:]) + eps   (one wave per column)
// ---------------------------------------------------------------------------
__global__ __launch_bounds__(256) void z_kernel(
    const __bf16* __restrict__ qbT, const float* __restrict__ kmean,
    float* __restrict__ z)
{
    const int b    = blockIdx.y;
    const int w    = threadIdx.x >> 5;
    const int lane = threadIdx.x & 31;
    const int n    = blockIdx.x * 8 + w;
    const __bf16* qrow = qbT + ((size_t)b * NPix + n) * DIM;
    const float*  km   = kmean + (size_t)b * DIM;
    float p = 0.f;
#pragma unroll
    for (int j = 0; j < 16; ++j) {
        int c = lane + 32 * j;
        p += (float)qrow[c] * km[c];
    }
    for (int off = 16; off > 0; off >>= 1) p += __shfl_xor(p, off, 32);
    if (lane == 0) z[(size_t)b * NPix + n] = p + 5e-4f;
}

// ---------------------------------------------------------------------------
// Per (b,n): attn/z, RMSNorm over 512 channels, * attn_norm_w * silu(g),
// write bf16 transposed (ready as Bt operand of the proj GEMM).
// ---------------------------------------------------------------------------
__global__ __launch_bounds__(256) void rms_gate_kernel(
    const float* __restrict__ attnT, const float* __restrict__ z,
    const float* __restrict__ normw, const __bf16* __restrict__ gT,
    __bf16* __restrict__ ogT)
{
    __shared__ float red[256];
    const int n = blockIdx.x;
    const int b = blockIdx.y;
    const size_t base = ((size_t)b * NPix + n) * DIM;
    const float invz = 1.f / z[(size_t)b * NPix + n];
    const int t = threadIdx.x;

    float a0 = attnT[base + t]       * invz;
    float a1 = attnT[base + t + 256] * invz;
    red[t] = a0 * a0 + a1 * a1;
    __syncthreads();
    for (int st = 128; st > 0; st >>= 1) {
        if (t < st) red[t] += red[t + st];
        __syncthreads();
    }
    float r = rsqrtf(red[0] * (1.f / (float)DIM) + 1e-6f);
    ogT[base + t]       = (__bf16)(a0 * r * normw[t]       * (float)gT[base + t]);
    ogT[base + t + 256] = (__bf16)(a1 * r * normw[t + 256] * (float)gT[base + t + 256]);
}

// ---------------------------------------------------------------------------
// Final BN apply -> fp32 output (B,512,4096)
// ---------------------------------------------------------------------------
__global__ void apply_bn_out_kernel(
    const __bf16* __restrict__ pr, const float* __restrict__ coef,
    float* __restrict__ out)
{
    size_t idx = (size_t)blockIdx.x * 256 + threadIdx.x;
    int m = (int)((idx >> 12) & 511);
    out[idx] = coef[2 * m] * (float)pr[idx] + coef[2 * m + 1];
}

// ---------------------------------------------------------------------------
// Launch
// ---------------------------------------------------------------------------
extern "C" void kernel_launch(void* const* d_in, const int* in_sizes, int n_in,
                              void* d_out, int out_size, void* d_ws, size_t ws_size,
                              hipStream_t stream)
{
    const float* x          = (const float*)d_in[0];
    const float* qkvg_w     = (const float*)d_in[1];
    const float* qkvg_gamma = (const float*)d_in[2];
    const float* qkvg_beta  = (const float*)d_in[3];
    const float* attn_normw = (const float*)d_in[4];
    const float* proj_w     = (const float*)d_in[5];
    const float* proj_gamma = (const float*)d_in[6];
    const float* proj_beta  = (const float*)d_in[7];
    float* out = (float*)d_out;

    char* ws = (char*)d_ws;
    // ---- workspace layout (bytes) ----
    const size_t SZ_WB   = (size_t)M4 * DIM * 2;           // 2 MB
    const size_t SZ_PWB  = (size_t)DIM * DIM * 2;          // 0.5 MB
    const size_t SZ_NT   = (size_t)BATCH * NPix * DIM * 2; // 33.55 MB (bf16 B,N,512)
    const size_t SZ_Y    = (size_t)BATCH * M4 * NPix * 2;  // 134.2 MB
    const size_t OFF_WB   = 0;
    const size_t OFF_PWB  = OFF_WB  + SZ_WB;
    const size_t OFF_XBT  = OFF_PWB + SZ_PWB;
    const size_t OFF_Y    = OFF_XBT + SZ_NT;
    // Y region reused after activation kernel:
    const size_t OFF_ATT  = OFF_Y;                                   // fp32 (B,N,512) 67.1 MB
    const size_t OFF_OGT  = OFF_Y + (size_t)BATCH * NPix * DIM * 4;  // bf16 33.55 MB
    const size_t OFF_PR   = OFF_OGT + SZ_NT;                         // bf16 33.55 MB
    const size_t OFF_QBT  = OFF_Y + SZ_Y;
    const size_t OFF_KB   = OFF_QBT + SZ_NT;
    const size_t OFF_VB   = OFF_KB  + SZ_NT;
    const size_t OFF_GT   = OFF_VB  + SZ_NT;
    const size_t OFF_VK   = OFF_GT  + SZ_NT;                         // bf16 (B,512,512)
    const size_t OFF_KM   = OFF_VK  + (size_t)BATCH * DIM * DIM * 2;
    const size_t OFF_Z    = OFF_KM  + (size_t)BATCH * DIM * 4;
    const size_t OFF_ST1  = OFF_Z   + (size_t)BATCH * NPix * 4;
    const size_t OFF_CF1  = OFF_ST1 + (size_t)M4 * 2 * 4;
    const size_t OFF_ST2  = OFF_CF1 + (size_t)M4 * 2 * 4;
    const size_t OFF_CF2  = OFF_ST2 + (size_t)DIM * 2 * 4;
    (void)OFF_CF2;

    __bf16* wb   = (__bf16*)(ws + OFF_WB);
    __bf16* pwb  = (__bf16*)(ws + OFF_PWB);
    __bf16* xbT  = (__bf16*)(ws + OFF_XBT);
    __bf16* y    = (__bf16*)(ws + OFF_Y);
    float*  attnT= (float*) (ws + OFF_ATT);
    __bf16* ogT  = (__bf16*)(ws + OFF_OGT);
    __bf16* pr   = (__bf16*)(ws + OFF_PR);
    __bf16* qbT  = (__bf16*)(ws + OFF_QBT);
    __bf16* kb   = (__bf16*)(ws + OFF_KB);
    __bf16* vb   = (__bf16*)(ws + OFF_VB);
    __bf16* gT   = (__bf16*)(ws + OFF_GT);
    __bf16* vk   = (__bf16*)(ws + OFF_VK);
    float*  km   = (float*) (ws + OFF_KM);
    float*  zbuf = (float*) (ws + OFF_Z);
    float*  st1  = (float*) (ws + OFF_ST1);
    float*  cf1  = (float*) (ws + OFF_CF1);
    float*  st2  = (float*) (ws + OFF_ST2);
    float*  cf2  = (float*) (ws + OFF_CF2);

    const float invCnt = 1.f / (float)(BATCH * NPix);

    // 0) weight conversions
    f2bf_kernel<<<(M4 * DIM + 255) / 256, 256, 0, stream>>>(qkvg_w, wb, M4 * DIM);
    f2bf_kernel<<<(DIM * DIM + 255) / 256, 256, 0, stream>>>(proj_w, pwb, DIM * DIM);

    // 1) transpose x -> xbT (B,N,512) bf16
    transpose_x_kernel<<<dim3(NPix / 32, DIM / 32, BATCH), 256, 0, stream>>>(x, xbT);

    // 2) QKVG GEMM: y = W(2048x512) @ X(512x4096)   (A shared across batch)
    gemm_bf16_wmma<0><<<dim3(NPix / 128, M4 / 128, BATCH), 256, 0, stream>>>(
        wb, 0, xbT, (long long)NPix * DIM, y, (long long)M4 * NPix,
        DIM, NPix, 1.0f);

    // 3) BN1 stats + finalize
    bn_stats_kernel<<<M4, 256, 0, stream>>>(y, st1, (long long)M4 * NPix);
    bn_finalize_kernel<<<(M4 + 255) / 256, 256, 0, stream>>>(
        st1, qkvg_gamma, qkvg_beta, cf1, M4, invCnt);

    // 4) BN apply + activations + k-mean
    act_kernel<<<dim3(M4, BATCH), 256, 0, stream>>>(y, cf1, qbT, kb, vb, gT, km);

    // 5) vk = (v*s)@(k*s)^T : (512x4096)@(4096x512), scale^2 = 1/N
    gemm_bf16_wmma<0><<<dim3(DIM / 128, DIM / 128, BATCH), 256, 0, stream>>>(
        vb, (long long)DIM * NPix, kb, (long long)DIM * NPix,
        vk, (long long)DIM * DIM, NPix, DIM, 1.0f / (float)NPix);

    // 6) z = kmean . q + eps
    z_kernel<<<dim3(NPix / 8, BATCH), 256, 0, stream>>>(qbT, km, zbuf);

    // 7) attn = vk @ q : (512x512)@(512x4096) -> fp32 transposed (B,N,512)
    gemm_bf16_wmma<1><<<dim3(NPix / 128, DIM / 128, BATCH), 256, 0, stream>>>(
        vk, (long long)DIM * DIM, qbT, (long long)NPix * DIM,
        attnT, (long long)NPix * DIM, DIM, DIM, 1.0f);

    // 8) /z, RMSNorm over channels, *norm_w, *silu(g) -> ogT bf16 (B,N,512)
    rms_gate_kernel<<<dim3(NPix, BATCH), 256, 0, stream>>>(
        attnT, zbuf, attn_normw, gT, ogT);

    // 9) proj GEMM: pr = Wp(512x512) @ o(512x4096)
    gemm_bf16_wmma<0><<<dim3(NPix / 128, DIM / 128, BATCH), 256, 0, stream>>>(
        pwb, 0, ogT, (long long)NPix * DIM, pr, (long long)DIM * NPix,
        DIM, NPix, 1.0f);

    // 10) BN2 stats + finalize + apply -> fp32 out
    bn_stats_kernel<<<DIM, 256, 0, stream>>>(pr, st2, (long long)DIM * NPix);
    bn_finalize_kernel<<<(DIM + 255) / 256, 256, 0, stream>>>(
        st2, proj_gamma, proj_beta, cf2, DIM, invCnt);
    apply_bn_out_kernel<<<(BATCH * DIM * NPix) / 256, 256, 0, stream>>>(pr, cf2, out);
}
